// MyModel_80771154969332
// MI455X (gfx1250) — matrix operand order repaired and checked by reading, not compiled, not run on hardware
//
#include <hip/hip_runtime.h>

// ---------------------------------------------------------------------------
// CDNA5 (gfx1250) fused 2-layer BiLSTM + max-pool + linear + log_softmax
// ---------------------------------------------------------------------------

typedef __attribute__((ext_vector_type(16))) __bf16 v16bf;
typedef __attribute__((ext_vector_type(8)))  float  v8f;
typedef __attribute__((ext_vector_type(8)))  unsigned int v8u;

#define S1   1024   // sentences
#define T1   128    // words per sentence
#define D1   343    // word feature dim
#define H1   150    // hidden per direction
#define KX   352    // D1 padded to 32
#define KH   160    // H1 padded to 32
#define KT1  512    // KX + KH
#define NP1  640    // 4*H1 (=600) padded to 16*40
#define KE   320    // 2*H1 padded (sent_emb cols, 150|pad|150|pad)
#define KT2  480    // KE + KH (layer-2 combined K)
#define NP2  1280   // 2 dirs * NP1

// ---- workspace layout (bytes) ----
#define OFF_W1   0u            // [2][NP1][KT1] bf16   = 1,310,720
#define OFF_B1   1310720u      // [2][NP1] f32         = 5,120
#define OFF_W2   1315840u      // [NP2][KT2] bf16      = 1,228,800
#define OFF_B2   2544640u      // [NP2] f32            = 5,120
#define OFF_SE   2549760u      // [S1][KE] bf16        = 655,360
#define OFF_XG2  3205120u      // [S1][NP2] f32        = 5,242,880
#define OFF_H2   8448000u      // [S1][300] f32        = 1,228,800

__device__ __forceinline__ unsigned short f32_to_bf16(float f) {
    unsigned u = __builtin_bit_cast(unsigned, f);
    u += 0x7FFFu + ((u >> 16) & 1u);          // round-to-nearest-even
    return (unsigned short)(u >> 16);
}
__device__ __forceinline__ float bf16_to_f32(unsigned short s) {
    return __builtin_bit_cast(float, (unsigned)s << 16);
}
__device__ __forceinline__ float sigmoidf_(float x) {
    return 1.0f / (1.0f + __expf(-x));
}
__device__ __forceinline__ float tanhf_(float x) {
    float xc = fminf(fmaxf(x, -15.f), 15.f);
    float e = __expf(2.f * xc);
    return (e - 1.f) / (e + 1.f);
}
__device__ __forceinline__ v8f vzero8() {
    v8f v;
#pragma unroll
    for (int i = 0; i < 8; ++i) v[i] = 0.f;
    return v;
}
// pack two 16B chunks into one v16bf fragment
__device__ __forceinline__ v16bf load_frag(const unsigned short* p0,
                                           const unsigned short* p1) {
    uint4 a = *(const uint4*)p0;
    uint4 b = *(const uint4*)p1;
    v8u u;
    u[0] = a.x; u[1] = a.y; u[2] = a.z; u[3] = a.w;
    u[4] = b.x; u[5] = b.y; u[6] = b.z; u[7] = b.w;
    return __builtin_bit_cast(v16bf, u);
}

// ---------------------------------------------------------------------------
// prep kernels: build padded bf16 weight images + combined biases
// ---------------------------------------------------------------------------
__global__ void prep_w1(const float* __restrict__ wihf, const float* __restrict__ whhf,
                        const float* __restrict__ wihb, const float* __restrict__ whhb,
                        unsigned short* __restrict__ W1) {
    int idx = blockIdx.x * 256 + threadIdx.x;
    if (idx >= 2 * NP1 * KT1) return;
    int dir = idx / (NP1 * KT1);
    int rem = idx - dir * NP1 * KT1;
    int n = rem / KT1, k = rem - n * KT1;
    const float* wih = dir ? wihb : wihf;
    const float* whh = dir ? whhb : whhf;
    float v = 0.f;
    if (n < 600) {
        if (k < D1)                      v = wih[n * D1 + k];
        else if (k >= KX && k < KX + H1) v = whh[n * H1 + (k - KX)];
    }
    W1[idx] = f32_to_bf16(v);
}

__global__ void prep_b1(const float* __restrict__ bihf, const float* __restrict__ bhhf,
                        const float* __restrict__ bihb, const float* __restrict__ bhhb,
                        float* __restrict__ B1) {
    int idx = blockIdx.x * 256 + threadIdx.x;
    if (idx >= 2 * NP1) return;
    int dir = idx / NP1, n = idx - dir * NP1;
    float v = 0.f;
    if (n < 600) v = (dir ? bihb[n] + bhhb[n] : bihf[n] + bhhf[n]);
    B1[idx] = v;
}

__global__ void prep_w2(const float* __restrict__ wihf, const float* __restrict__ whhf,
                        const float* __restrict__ wihb, const float* __restrict__ whhb,
                        unsigned short* __restrict__ W2) {
    int idx = blockIdx.x * 256 + threadIdx.x;
    if (idx >= NP2 * KT2) return;
    int n = idx / KT2, k = idx - n * KT2;
    int dir = (n >= NP1) ? 1 : 0;
    int nn = n - dir * NP1;
    const float* wih = dir ? wihb : wihf;   // [600][300]
    const float* whh = dir ? whhb : whhf;   // [600][150]
    float v = 0.f;
    if (nn < 600) {
        if (k < H1)                      v = wih[nn * 300 + k];               // fwd half
        else if (k >= 160 && k < 310)    v = wih[nn * 300 + 150 + (k - 160)]; // bwd half
        else if (k >= KE && k < KE + H1) v = whh[nn * H1 + (k - KE)];         // recurrent
    }
    W2[idx] = f32_to_bf16(v);
}

__global__ void prep_b2(const float* __restrict__ bihf, const float* __restrict__ bhhf,
                        const float* __restrict__ bihb, const float* __restrict__ bhhb,
                        float* __restrict__ B2) {
    int idx = blockIdx.x * 256 + threadIdx.x;
    if (idx >= NP2) return;
    int dir = (idx >= NP1) ? 1 : 0;
    int nn = idx - dir * NP1;
    float v = 0.f;
    if (nn < 600) v = (dir ? bihb[nn] + bhhb[nn] : bihf[nn] + bhhf[nn]);
    B2[idx] = v;
}

// ---------------------------------------------------------------------------
// Layer-1 BiLSTM: one workgroup = 16 sentences x 1 direction, full time loop.
// Per step: gates[16,640] = [x_t | h] (16x512 bf16, LDS) x W1^T (WMMA bf16),
// then fused sigmoid/tanh cell update + running max of h (never materializes
// word_out). Outputs sent_emb bf16 [1024][320].
// ---------------------------------------------------------------------------
__global__ void __launch_bounds__(256)
lstm1_kernel(const float* __restrict__ x,             // [S1][T1][D1]
             const unsigned short* __restrict__ W1,   // [2][NP1][KT1]
             const float* __restrict__ B1,            // [2][NP1]
             unsigned short* __restrict__ semb) {     // [S1][KE]
    __shared__ __align__(16) unsigned short Abuf[16 * KT1]; // 16 KB
    __shared__ float Gbuf[16 * NP1];                        // 40 KB
    const int tid  = threadIdx.x;
    const int lane = tid & 31;
    const int wave = tid >> 5;
    const int sb   = blockIdx.x;   // sentence block 0..63
    const int dir  = blockIdx.y;   // 0=fwd 1=bwd
    const unsigned short* Wd = W1 + (size_t)dir * NP1 * KT1;
    const float* bias = B1 + dir * NP1;

    for (int i = tid; i < 16 * KT1; i += 256) Abuf[i] = 0;  // zeros h + pads

    float c_reg[10], hmax[10];
#pragma unroll
    for (int it = 0; it < 10; ++it) { c_reg[it] = 0.f; hmax[it] = -1e30f; }

    const int arow = lane & 15;
    const int asel = (lane >> 4) ? 8 : 0;     // A-fragment half-lane K offset
    const int brow = lane & 15;
    const int bk   = (lane >> 4) * 8;         // B-fragment half-lane K offset

    for (int t = 0; t < T1; ++t) {
        const int te = dir ? (T1 - 1 - t) : t;
        // stage x_t (f32 -> bf16) into A buffer cols [0,343)
        for (int i = tid; i < 16 * D1; i += 256) {
            int m = i / D1, k = i - m * D1;
            float v = x[((size_t)(sb * 16 + m) * T1 + te) * D1 + k];
            Abuf[m * KT1 + k] = f32_to_bf16(v);
        }
        __syncthreads();

        // Opaque zero offset: blocks LICM from hoisting (and scratch-spilling)
        // the 80 B-fragments out of the time loop, while keeping the pointer's
        // kernarg provenance so the loads stay global_load_b128 (LOADcnt only),
        // not flat_load (LOADcnt+DScnt). Weight image is L2-resident.
        unsigned int toff = 0;
        asm volatile("" : "+s"(toff));
        const unsigned short* Wt = Wd + toff;

        // GEMM: 8 waves x 5 N-tiles, K = 16 chunks of 32
        v8f acc[5];
#pragma unroll
        for (int q = 0; q < 5; ++q) acc[q] = vzero8();
        for (int kc = 0; kc < 16; ++kc) {
            const unsigned short* ap = &Abuf[arow * KT1 + kc * 32 + asel];
            v16bf af = load_frag(ap, ap + 16);
#pragma unroll
            for (int q = 0; q < 5; ++q) {
                int nt = wave * 5 + q;
                const unsigned short* bp =
                    Wt + (size_t)(nt * 16 + brow) * KT1 + kc * 32 + bk;
                v16bf bf = load_frag(bp, bp + 16);
                acc[q] = __builtin_amdgcn_wmma_f32_16x16x32_bf16(
                    false, af, false, bf, (short)0, acc[q], false, false);
            }
        }
        // scatter accumulators to gate buffer
        const int mtop = (lane >> 4) * 8;
#pragma unroll
        for (int q = 0; q < 5; ++q) {
            int nb = (wave * 5 + q) * 16 + (lane & 15);
#pragma unroll
            for (int r = 0; r < 8; ++r)
                Gbuf[(mtop + r) * NP1 + nb] = acc[q][r];
        }
        __syncthreads();

        // fused LSTM cell update + running max, h -> A buffer cols [352,502)
#pragma unroll
        for (int it = 0; it < 10; ++it) {
            int p = tid + it * 256;
            if (p < 16 * H1) {
                int m = p / H1, j = p - m * H1;
                float gi = Gbuf[m * NP1 + j]          + bias[j];
                float gf = Gbuf[m * NP1 + H1 + j]     + bias[H1 + j];
                float gg = Gbuf[m * NP1 + 2 * H1 + j] + bias[2 * H1 + j];
                float go = Gbuf[m * NP1 + 3 * H1 + j] + bias[3 * H1 + j];
                float i_ = sigmoidf_(gi), f_ = sigmoidf_(gf), o_ = sigmoidf_(go);
                float c  = f_ * c_reg[it] + i_ * tanhf_(gg);
                float h  = o_ * tanhf_(c);
                c_reg[it] = c;
                hmax[it]  = fmaxf(hmax[it], h);
                Abuf[m * KT1 + KX + j] = f32_to_bf16(h);
            }
        }
        __syncthreads();
    }

    // emit max-pooled embedding (bf16) for this direction
#pragma unroll
    for (int it = 0; it < 10; ++it) {
        int p = tid + it * 256;
        if (p < 16 * H1) {
            int m = p / H1, j = p - m * H1;
            semb[(size_t)(sb * 16 + m) * KE + dir * 160 + j] = f32_to_bf16(hmax[it]);
        }
    }
    if (tid < 160) {  // zero pad cols 150..159 of this direction's half
        int m = tid / 10, j = H1 + tid % 10;
        semb[(size_t)(sb * 16 + m) * KE + dir * 160 + j] = 0;
    }
}

// ---------------------------------------------------------------------------
// Layer-2 input projection (batch-parallel part of the scan), WMMA GEMM:
// xg2[1024,1280] = sent_emb[1024,320] x W2(x-part)^T + bias2
// ---------------------------------------------------------------------------
__global__ void __launch_bounds__(256)
xg2_kernel(const unsigned short* __restrict__ semb, // [S1][KE] bf16
           const unsigned short* __restrict__ W2,   // [NP2][KT2] bf16
           const float* __restrict__ B2,            // [NP2]
           float* __restrict__ xg2) {               // [S1][NP2]
    __shared__ __align__(16) unsigned short Abuf[16 * KE]; // 10 KB
    const int tid  = threadIdx.x;
    const int lane = tid & 31;
    const int wave = tid >> 5;
    const int mt   = blockIdx.x;             // 0..63
    const int nt   = blockIdx.y * 8 + wave;  // 0..79

    const unsigned* src = (const unsigned*)(semb + (size_t)mt * 16 * KE);
    unsigned* dst = (unsigned*)Abuf;
    for (int i = tid; i < 16 * KE / 2; i += 256) dst[i] = src[i];
    __syncthreads();

    v8f acc = vzero8();
    const int arow = lane & 15;
    const int asel = (lane >> 4) ? 8 : 0;
    const int brow = nt * 16 + (lane & 15);
    const int bk   = (lane >> 4) * 8;
    for (int kc = 0; kc < KE / 32; ++kc) {
        const unsigned short* ap = &Abuf[arow * KE + kc * 32 + asel];
        v16bf af = load_frag(ap, ap + 16);
        const unsigned short* bp = W2 + (size_t)brow * KT2 + kc * 32 + bk;
        v16bf bf = load_frag(bp, bp + 16);
        acc = __builtin_amdgcn_wmma_f32_16x16x32_bf16(
            false, af, false, bf, (short)0, acc, false, false);
    }
    float bcol = B2[nt * 16 + (lane & 15)];
    const int mtop = (lane >> 4) * 8;
#pragma unroll
    for (int r = 0; r < 8; ++r)
        xg2[(size_t)(mt * 16 + mtop + r) * NP2 + nt * 16 + (lane & 15)] = acc[r] + bcol;
}

// ---------------------------------------------------------------------------
// Layer-2 serial recurrence (batch=1, 1024 steps). One workgroup per
// direction, 608 threads: thread n computes gate n via 150-MAC dot with
// h (LDS); threads j<150 apply the cell update. The 75-dword Whh2 row is
// loop-invariant and gets register-cached by the compiler (intended).
// ---------------------------------------------------------------------------
__global__ void __launch_bounds__(608)
lstm2_kernel(const float* __restrict__ xg2,          // [S1][NP2]
             const unsigned short* __restrict__ W2,  // [NP2][KT2]
             float* __restrict__ h2out) {            // [S1][300]
    __shared__ float hbuf[152];
    __shared__ float gbuf[608];
    const int n = threadIdx.x;
    const int dir = blockIdx.x;
    if (n < 152) hbuf[n] = 0.f;
    float c = 0.f;
    const unsigned* wrow =
        (const unsigned*)(W2 + (size_t)(dir * NP1 + n) * KT2 + KE); // 75 dwords
    __syncthreads();

    for (int s = 0; s < S1; ++s) {
        const int row = dir ? (S1 - 1 - s) : s;
        if (n < 600) {
            float acc = xg2[(size_t)row * NP2 + dir * NP1 + n];
            for (int k2 = 0; k2 < H1 / 2; ++k2) {
                unsigned u = wrow[k2];
                acc += bf16_to_f32((unsigned short)(u & 0xFFFFu)) * hbuf[2 * k2];
                acc += bf16_to_f32((unsigned short)(u >> 16))     * hbuf[2 * k2 + 1];
            }
            gbuf[n] = acc;
        }
        __syncthreads();
        if (n < H1) {
            float i_ = sigmoidf_(gbuf[n]);
            float f_ = sigmoidf_(gbuf[n + H1]);
            float g_ = tanhf_(gbuf[n + 2 * H1]);
            float o_ = sigmoidf_(gbuf[n + 3 * H1]);
            c = f_ * c + i_ * g_;
            float h = o_ * tanhf_(c);
            hbuf[n] = h;
            h2out[(size_t)row * 300 + dir * H1 + n] = h;
        }
        __syncthreads();
    }
}

// ---------------------------------------------------------------------------
// Head: logits = sent_out @ w_out^T + b_out; log_softmax over 7 classes.
// ---------------------------------------------------------------------------
__global__ void __launch_bounds__(256)
head_kernel(const float* __restrict__ h2out, // [S1][300]
            const float* __restrict__ wout,  // [7][300]
            const float* __restrict__ bout,  // [7]
            float* __restrict__ out) {       // [S1][7]
    int row = blockIdx.x * 256 + threadIdx.x;
    if (row >= S1) return;
    const float* hr = h2out + (size_t)row * 300;
    float l[7];
    float mx = -1e30f;
#pragma unroll
    for (int cix = 0; cix < 7; ++cix) {
        float a = bout[cix];
        const float* w = wout + cix * 300;
        for (int k = 0; k < 300; ++k) a += w[k] * hr[k];
        l[cix] = a;
        mx = fmaxf(mx, a);
    }
    float ssum = 0.f;
#pragma unroll
    for (int cix = 0; cix < 7; ++cix) ssum += __expf(l[cix] - mx);
    float lse = __logf(ssum) + mx;
#pragma unroll
    for (int cix = 0; cix < 7; ++cix) out[row * 7 + cix] = l[cix] - lse;
}

// ---------------------------------------------------------------------------
extern "C" void kernel_launch(void* const* d_in, const int* in_sizes, int n_in,
                              void* d_out, int out_size, void* d_ws, size_t ws_size,
                              hipStream_t stream) {
    const float* x       = (const float*)d_in[0];
    const float* w_ih_f1 = (const float*)d_in[1];
    const float* w_hh_f1 = (const float*)d_in[2];
    const float* b_ih_f1 = (const float*)d_in[3];
    const float* b_hh_f1 = (const float*)d_in[4];
    const float* w_ih_b1 = (const float*)d_in[5];
    const float* w_hh_b1 = (const float*)d_in[6];
    const float* b_ih_b1 = (const float*)d_in[7];
    const float* b_hh_b1 = (const float*)d_in[8];
    const float* w_ih_f2 = (const float*)d_in[9];
    const float* w_hh_f2 = (const float*)d_in[10];
    const float* b_ih_f2 = (const float*)d_in[11];
    const float* b_hh_f2 = (const float*)d_in[12];
    const float* w_ih_b2 = (const float*)d_in[13];
    const float* w_hh_b2 = (const float*)d_in[14];
    const float* b_ih_b2 = (const float*)d_in[15];
    const float* b_hh_b2 = (const float*)d_in[16];
    const float* w_out   = (const float*)d_in[17];
    const float* b_out   = (const float*)d_in[18];
    float* out = (float*)d_out;

    char* ws = (char*)d_ws;
    unsigned short* W1   = (unsigned short*)(ws + OFF_W1);
    float*          B1   = (float*)(ws + OFF_B1);
    unsigned short* W2   = (unsigned short*)(ws + OFF_W2);
    float*          B2   = (float*)(ws + OFF_B2);
    unsigned short* SEMB = (unsigned short*)(ws + OFF_SE);
    float*          XG2  = (float*)(ws + OFF_XG2);
    float*          H2   = (float*)(ws + OFF_H2);

    prep_w1<<<(2 * NP1 * KT1 + 255) / 256, 256, 0, stream>>>(
        w_ih_f1, w_hh_f1, w_ih_b1, w_hh_b1, W1);
    prep_b1<<<(2 * NP1 + 255) / 256, 256, 0, stream>>>(
        b_ih_f1, b_hh_f1, b_ih_b1, b_hh_b1, B1);
    prep_w2<<<(NP2 * KT2 + 255) / 256, 256, 0, stream>>>(
        w_ih_f2, w_hh_f2, w_ih_b2, w_hh_b2, W2);
    prep_b2<<<(NP2 + 255) / 256, 256, 0, stream>>>(
        b_ih_f2, b_hh_f2, b_ih_b2, b_hh_b2, B2);

    lstm1_kernel<<<dim3(S1 / 16, 2), 256, 0, stream>>>(x, W1, B1, SEMB);
    xg2_kernel<<<dim3(S1 / 16, NP2 / 128), 256, 0, stream>>>(SEMB, W2, B2, XG2);
    lstm2_kernel<<<2, 608, 0, stream>>>(XG2, W2, H2);
    head_kernel<<<(S1 + 255) / 256, 256, 0, stream>>>(H2, w_out, b_out, out);
}
